// Quantize_39591008535131
// MI455X (gfx1250) — matrix operand order repaired and checked by reading, not compiled, not run on hardware
//
#include <hip/hip_runtime.h>
#include <hip/hip_bf16.h>

// ---------------- problem constants (from reference) ----------------
#define NBOOK   4
#define CODES   1024
#define DIM     64
#define THW     8192                   // T*H*W = 8*32*32
#define NPTS    32768                  // B*T*H*W
#define ZBSTRIDE 2097152               // (NB*D)*THW = 256*8192
#define ZELEMS  8388608                // B*(NB*D)*T*H*W
#define NTILES  2048                   // NPTS/16
#define CTILES  64                     // CODES/16
#define ENC_OFF  8388608
#define LOSS_OFF 8519680               // ENC_OFF + NPTS*NBOOK
#define PERP_OFF 8519681
#define NWAVES   4                     // waves per workgroup
#define TPW      2                     // point-tiles per wave (M=32)
#define NPARTIAL 4096                  // total waves in main grid

typedef __attribute__((ext_vector_type(16))) __bf16 bf16x16;
typedef __attribute__((ext_vector_type(8)))  float  f32x8;

// float -> bf16 (RNE) via bit ops: only needs 2-byte storage + bit_cast.
__device__ __forceinline__ unsigned short f2bf_bits(float f) {
    unsigned u = __builtin_bit_cast(unsigned, f);
    unsigned r = u + 0x7FFFu + ((u >> 16) & 1u);
    return (unsigned short)(r >> 16);
}
__device__ __forceinline__ float bfbits2f(unsigned short h) {
    return __builtin_bit_cast(float, (unsigned)h << 16);
}

// ---------------------------------------------------------------------
// Kernel 1: stage embeddings as one combined bf16 panel in WMMA-B lane
// layout, one contiguous 4KB block per (book, code-tile):
//   panel[k][ct][v][L][j], v: 0=hi K0-31, 1=hi K32-63, 2=lo K0-31, 3=lo K32-63
//   lane L: column n = L%16, K-half = L/16; element j -> K = 16*(L/16)+j.
// Also computes ||e||^2 per code.
// ---------------------------------------------------------------------
__global__ void vq_prep(const float* __restrict__ emb,
                        unsigned short* __restrict__ panel,
                        float* __restrict__ e2) {
    int g = blockIdx.x * blockDim.x + threadIdx.x;   // one thread per (k, code)
    if (g >= NBOOK * CODES) return;
    int k = g >> 10, c = g & (CODES - 1);
    int tile = c >> 4, n = c & 15;
    const float* ev = emb + (size_t)(k * CODES + c) * DIM;
    float s = 0.f;
    for (int d = 0; d < DIM; ++d) {
        float v = ev[d];
        s += v * v;
        int q = d >> 5, rem = d & 31;
        int L = (rem >> 4) * 16 + n;
        int j = rem & 15;
        size_t base = (size_t)(k * CTILES + tile) * 2048;   // shorts per ctile block
        unsigned short hs = f2bf_bits(v);
        panel[base + (size_t)(q      ) * 512 + L * 16 + j] = hs;
        panel[base + (size_t)(q + 2  ) * 512 + L * 16 + j] =
            f2bf_bits(v - bfbits2f(hs));
    }
    e2[g] = s;
}

// ---------------------------------------------------------------------
// Kernel 2: 128-thread workgroup (4 waves), each wave owns 2 point-tiles
// (M=32) of one book. B code-tiles are double-buffered in LDS, filled
// with gfx1250 async-to-LDS copies shared by all 4 waves.
// bf16x3 compensated GEMM -> fused argmin -> outputs.
// ---------------------------------------------------------------------
__global__ void __launch_bounds__(128)
vq_main(const float* __restrict__ z, const float* __restrict__ emb,
        const unsigned short* __restrict__ panel,
        const float* __restrict__ e2,
        unsigned* __restrict__ counts,
        float* __restrict__ partials,
        float* __restrict__ out) {
    __shared__ bf16x16 sB[2][4][32];     // [buffer][vector][lane] : 2 x 4KB
    __shared__ int sidx[NWAVES][32];

    const int k     = blockIdx.x >> 8;             // book
    const int btile = (blockIdx.x & 255) * (NWAVES * TPW);
    const int tid   = threadIdx.x;
    const int w     = tid >> 5;                    // wave in workgroup
    const int lane  = tid & 31;
    const int lm    = lane & 15;                   // A row / D column index
    const int lg    = lane >> 4;                   // K-half group

    const int pt0 = btile + w * TPW;               // first point-tile of wave
    const int n0  = pt0 * 16;                      // 32 consecutive points
    const int b   = n0 / THW;                      // 8192 % 32 == 0
    const int thw0 = n0 % THW;

    // ---- load A (2 tiles x 16 points x 64 dims), hi/lo bf16 split ----
    // A k order used here (contiguous, same as B panel): lane holds row m=lm; element j is
    // K = 16*lg + j; chunk q adds 32.
    bf16x16 ahi[TPW][2], alo[TPW][2];
#pragma unroll
    for (int t = 0; t < TPW; ++t) {
        const size_t zb = (size_t)b * ZBSTRIDE + (size_t)k * DIM * THW
                        + thw0 + t * 16 + lm;
#pragma unroll
        for (int q = 0; q < 2; ++q) {
#pragma unroll
            for (int j = 0; j < 16; ++j) {
                int d = q * 32 + 16 * lg + j;
                float v = z[zb + (size_t)d * THW];
                unsigned short hs = f2bf_bits(v);
                ahi[t][q][j] = __builtin_bit_cast(__bf16, hs);
                alo[t][q][j] = __builtin_bit_cast(__bf16, f2bf_bits(v - bfbits2f(hs)));
            }
        }
    }

    // async-copy helper state: thread tid fills bytes [tid*32, tid*32+32)
    const unsigned long long gpanel =
        (unsigned long long)(uintptr_t)panel
        + (unsigned long long)(k * CTILES) * 4096ull + (unsigned)tid * 32u;
    const unsigned ldsbase = (unsigned)(uintptr_t)(&sB[0][0][0]) + (unsigned)tid * 32u;

    // preload code-tile 0 into buffer 0 (2 x b128 async per thread-lane)
    {
        unsigned long long ga = gpanel;            // ct = 0
        unsigned la = ldsbase;                     // buffer 0
        asm volatile("global_load_async_to_lds_b128 %0, %1, off\n\t"
                     "global_load_async_to_lds_b128 %0, %1, off offset:16"
                     :: "v"(la), "v"(ga) : "memory");
    }

    const float* e2k = e2 + k * CODES;
    float minv[TPW][8];
    int   mini[TPW][8];
#pragma unroll
    for (int t = 0; t < TPW; ++t)
#pragma unroll
        for (int r = 0; r < 8; ++r) { minv[t][r] = 3.4e38f; mini[t][r] = 0; }

    // ---- sweep all 64 code tiles, double-buffered through LDS ----
    for (int ct = 0; ct < CTILES; ++ct) {
        __syncthreads();   // everyone finished reading buffer (ct+1)&1 last iter
        if (ct + 1 < CTILES) {
            unsigned long long ga = gpanel + (unsigned long long)(ct + 1) * 4096ull;
            unsigned la = ldsbase + (unsigned)((ct + 1) & 1) * 4096u;
            asm volatile("global_load_async_to_lds_b128 %0, %1, off\n\t"
                         "global_load_async_to_lds_b128 %0, %1, off offset:16"
                         :: "v"(la), "v"(ga) : "memory");
            asm volatile("s_wait_asynccnt 0x2" ::: "memory"); // current buf done
        } else {
            asm volatile("s_wait_asynccnt 0x0" ::: "memory");
        }
        __syncthreads();   // whole workgroup's copies for this buffer landed

        const int nb = ct & 1;
        bf16x16 bh0 = sB[nb][0][lane];
        bf16x16 bh1 = sB[nb][1][lane];
        bf16x16 bl0 = sB[nb][2][lane];
        bf16x16 bl1 = sB[nb][3][lane];

        int cidx = ct * 16 + lm;
        float ev2 = e2k[cidx];

#pragma unroll
        for (int t = 0; t < TPW; ++t) {
            f32x8 acc = {};
            acc = __builtin_amdgcn_wmma_f32_16x16x32_bf16(false, ahi[t][0], false, bh0,
                                                          (short)0, acc, false, false);
            acc = __builtin_amdgcn_wmma_f32_16x16x32_bf16(false, ahi[t][1], false, bh1,
                                                          (short)0, acc, false, false);
            acc = __builtin_amdgcn_wmma_f32_16x16x32_bf16(false, ahi[t][0], false, bl0,
                                                          (short)0, acc, false, false);
            acc = __builtin_amdgcn_wmma_f32_16x16x32_bf16(false, ahi[t][1], false, bl1,
                                                          (short)0, acc, false, false);
            acc = __builtin_amdgcn_wmma_f32_16x16x32_bf16(false, alo[t][0], false, bh0,
                                                          (short)0, acc, false, false);
            acc = __builtin_amdgcn_wmma_f32_16x16x32_bf16(false, alo[t][1], false, bh1,
                                                          (short)0, acc, false, false);
            // D layout: lane holds rows m = r + 8*lg, column c = ct*16 + lm
#pragma unroll
            for (int r = 0; r < 8; ++r) {
                float score = ev2 - 2.0f * acc[r];
                if (score < minv[t][r]) { minv[t][r] = score; mini[t][r] = cidx; }
            }
        }
    }

    // ---- argmin across the 16 lanes of each K-half group ----
#pragma unroll
    for (int off = 1; off < 16; off <<= 1) {
#pragma unroll
        for (int t = 0; t < TPW; ++t)
#pragma unroll
            for (int r = 0; r < 8; ++r) {
                float ov = __shfl_xor(minv[t][r], off);
                int   oi = __shfl_xor(mini[t][r], off);
                if (ov < minv[t][r] || (ov == minv[t][r] && oi < mini[t][r])) {
                    minv[t][r] = ov; mini[t][r] = oi;
                }
            }
    }
    // rows 0..7 of each tile live in lanes 0-15, rows 8..15 in lanes 16-31
    if (lane == 0) {
#pragma unroll
        for (int t = 0; t < TPW; ++t)
            for (int r = 0; r < 8; ++r) sidx[w][t * 16 + r] = mini[t][r];
    }
    if (lane == 16) {
#pragma unroll
        for (int t = 0; t < TPW; ++t)
            for (int r = 0; r < 8; ++r) sidx[w][t * 16 + 8 + r] = mini[t][r];
    }
    __syncthreads();

    const int myidx = sidx[w][lane];
    // encodings (B,T,H,W,NB) flat = n*NB + k, as float
    out[ENC_OFF + (size_t)(n0 + lane) * NBOOK + k] = (float)myidx;
    // deterministic code histogram (integer atomics)
    atomicAdd(&counts[k * CODES + myidx], 1u);

    // ---- quantized scatter + commitment-loss partial ----
    // lane L owns point n0+L; loop dims (coalesced 128B across 32 lanes)
    const size_t wbase = (size_t)b * ZBSTRIDE + (size_t)k * DIM * THW + thw0 + lane;
    const float* eq = emb + (size_t)(k * CODES + myidx) * DIM;
    float lsum = 0.f;
    for (int d = 0; d < DIM; ++d) {
        float qv = eq[d];
        size_t idx = wbase + (size_t)d * THW;
        float diff = z[idx] - qv;
        lsum += diff * diff;
        out[idx] = qv;          // quantized_st == quantized numerically
    }
#pragma unroll
    for (int off = 1; off < 32; off <<= 1) lsum += __shfl_xor(lsum, off);
    if (lane == 0) partials[blockIdx.x * NWAVES + w] = lsum;
}

// ---------------------------------------------------------------------
// Kernel 3: fixed-order reductions -> loss & perplexity scalars.
// ---------------------------------------------------------------------
__global__ void vq_finalize(const unsigned* __restrict__ counts,
                            const float* __restrict__ partials,
                            float* __restrict__ out) {
    __shared__ float red[256];
    const int t = threadIdx.x;

    float s = 0.f;
    for (int i = t; i < NPARTIAL; i += 256) s += partials[i];
    red[t] = s; __syncthreads();
    for (int off = 128; off > 0; off >>= 1) {
        if (t < off) red[t] += red[t + off];
        __syncthreads();
    }
    float total = red[0];
    __syncthreads();

    float pp = 0.f;
    for (int k = 0; k < NBOOK; ++k) {
        float h = 0.f;
        for (int c = t; c < CODES; c += 256) {
            float p = (float)counts[k * CODES + c] * (1.0f / (float)NPTS);
            h += p * logf(p + 1e-10f);
        }
        red[t] = h; __syncthreads();
        for (int off = 128; off > 0; off >>= 1) {
            if (t < off) red[t] += red[t + off];
            __syncthreads();
        }
        if (t == 0) pp += expf(-red[0]);
        __syncthreads();
    }
    if (t == 0) {
        out[LOSS_OFF] = 0.25f * total / (float)ZELEMS;
        out[PERP_OFF] = pp / (float)NBOOK;
    }
}

// ---------------------------------------------------------------------
extern "C" void kernel_launch(void* const* d_in, const int* in_sizes, int n_in,
                              void* d_out, int out_size, void* d_ws, size_t ws_size,
                              hipStream_t stream) {
    const float* z   = (const float*)d_in[0];
    const float* emb = (const float*)d_in[1];
    float* out = (float*)d_out;

    // workspace layout (all offsets >= 32B aligned)
    char* ws = (char*)d_ws;
    unsigned short* panel = (unsigned short*)(ws);            // 1 MB
    float*    e2       = (float*)   (ws + 1048576);           // 16 KB
    unsigned* counts   = (unsigned*)(ws + 1064960);           // 16 KB
    float*    partials = (float*)   (ws + 1081344);           // 16 KB

    hipMemsetAsync(counts, 0, NBOOK * CODES * sizeof(unsigned), stream);
    vq_prep<<<(NBOOK * CODES + 255) / 256, 256, 0, stream>>>(emb, panel, e2);
    vq_main<<<NBOOK * (NTILES / (NWAVES * TPW)), NWAVES * 32, 0, stream>>>(
        z, emb, panel, e2, counts, partials, out);
    vq_finalize<<<1, 256, 0, stream>>>(counts, partials, out);
}
